// MultiHeadAttentionLayer_75453985456263
// MI455X (gfx1250) — matrix operand order repaired and checked by reading, not compiled, not run on hardware
//
#include <hip/hip_runtime.h>
#include <hip/hip_bf16.h>

#define NDIM   128          // IN_DIM
#define HD     128          // H*D
#define QKVW   384          // 3*H*D
#define NHEAD  8
#define DHEAD  16

typedef __bf16 v16bf __attribute__((ext_vector_type(16)));
typedef float  v8f   __attribute__((ext_vector_type(8)));

// ---- helpers -------------------------------------------------------------

__device__ __forceinline__ __bf16 f2bf(float f) {
    unsigned u = __float_as_uint(f);
    unsigned r = (u + 0x7FFFu + ((u >> 16) & 1u)) >> 16;   // round-nearest-even
    union { unsigned short s; __bf16 b; } cv;
    cv.s = (unsigned short)r;
    return cv.b;
}

// order-preserving float<->uint mapping (for atomicMax over signed floats)
__device__ __forceinline__ unsigned f2ord(float f) {
    unsigned u = __float_as_uint(f);
    return (u & 0x80000000u) ? ~u : (u | 0x80000000u);
}
__device__ __forceinline__ float ord2f(unsigned u) {
    return (u & 0x80000000u) ? __uint_as_float(u & 0x7FFFFFFFu)
                             : __uint_as_float(~u);
}

// ---- kernel 1: pack Wq|Wk|Wv -> bf16, transposed [col][k]; concat biases --

__global__ void mha_prep(const float* __restrict__ Wq, const float* __restrict__ Wk,
                         const float* __restrict__ Wv, const float* __restrict__ bq,
                         const float* __restrict__ bk, const float* __restrict__ bv,
                         __bf16* __restrict__ Wt, float* __restrict__ biasCat) {
    int idx = blockIdx.x * blockDim.x + threadIdx.x;      // 0 .. 384*128
    if (idx < QKVW * NDIM) {
        int col = idx >> 7;                               // 0..383
        int k   = idx & 127;
        const float* W = (col < HD) ? Wq : (col < 2 * HD) ? Wk : Wv;
        int c = col & 127;
        Wt[col * NDIM + k] = f2bf(W[k * HD + c]);
    }
    if (idx < QKVW) {
        float b = (idx < HD) ? bq[idx] : (idx < 2 * HD) ? bk[idx - HD] : bv[idx - 2 * HD];
        biasCat[idx] = b;
    }
}

// ---- kernel 2: init out=0, denom=0, smax=ordered(-inf) -------------------

__global__ void mha_init(float* __restrict__ out, float* __restrict__ denom,
                         unsigned* __restrict__ smax, int n) {
    long idx = (long)blockIdx.x * blockDim.x + threadIdx.x;
    if (idx < (long)n * HD) out[idx] = 0.0f;
    if (idx < (long)n * NHEAD) {
        denom[idx] = 0.0f;
        smax[idx]  = f2ord(-1e30f);
    }
}

// ---- kernel 3: QKV = h @ [Wq|Wk|Wv] + bias via v_wmma_f32_16x16x32_bf16 --
// block = 256 threads = 8 waves; block tile = 256 rows x 384 cols.
// Each wave owns TWO 16-row M-tiles so every B fragment feeds two WMMAs.
// QKV buffer is row-padded to a multiple of 256, so stores need NO bounds
// checks (no exec-mask churn in the hot loop); A loads clamp in the prolog.

__device__ __forceinline__ void load_afrag(const float* __restrict__ hrow,
                                           int kb, v16bf afrag[4]) {
#pragma unroll
    for (int ks = 0; ks < 4; ++ks) {
        const int k0 = ks * 32 + kb;
        const float4* p0 = (const float4*)(hrow + k0);
        const float4* p1 = (const float4*)(hrow + k0 + 8);
        float4 x0 = p0[0], x1 = p0[1];
        float4 y0 = p1[0], y1 = p1[1];
        afrag[ks][0]  = f2bf(x0.x); afrag[ks][1]  = f2bf(x0.y);
        afrag[ks][2]  = f2bf(x0.z); afrag[ks][3]  = f2bf(x0.w);
        afrag[ks][4]  = f2bf(x1.x); afrag[ks][5]  = f2bf(x1.y);
        afrag[ks][6]  = f2bf(x1.z); afrag[ks][7]  = f2bf(x1.w);
        afrag[ks][8]  = f2bf(y0.x); afrag[ks][9]  = f2bf(y0.y);
        afrag[ks][10] = f2bf(y0.z); afrag[ks][11] = f2bf(y0.w);
        afrag[ks][12] = f2bf(y1.x); afrag[ks][13] = f2bf(y1.y);
        afrag[ks][14] = f2bf(y1.z); afrag[ks][15] = f2bf(y1.w);
    }
}

__global__ __launch_bounds__(256) void mha_gemm_qkv(
        const float* __restrict__ h, const __bf16* __restrict__ Wt,
        const float* __restrict__ biasCat, float* __restrict__ QKV, int n) {
    const int lane = threadIdx.x & 31;
    const int wave = threadIdx.x >> 5;
    const int rowBase = blockIdx.x * 256 + wave * 32;     // two 16-row tiles

    // ---- A fragments: lane<16 -> M=lane&15, K in {kb..kb+15}
    const int m  = lane & 15;
    const int kb = (lane >> 4) * 16;
    int r0 = rowBase + m;          int lr0 = r0 < n ? r0 : (n - 1);
    int r1 = rowBase + 16 + m;     int lr1 = r1 < n ? r1 : (n - 1);

    v16bf afrag0[4], afrag1[4];
    load_afrag(h + (long)lr0 * NDIM, kb, afrag0);
    load_afrag(h + (long)lr1 * NDIM, kb, afrag1);

    // ---- B fragment addressing: lane = N, K split by lane half
    const int nn   = lane & 15;
    const int kb2  = (lane >> 4) * 16;
    const int rsub = (lane >> 4) * 8;                     // D-row offset per lane half

    for (int ct = 0; ct < 24; ++ct) {
        const int col = ct * 16 + nn;
        // gather all four B fragments first: one load clause, one wait,
        // then 8 back-to-back WMMAs on the XDL pipe.
        v16bf bf[4];
#pragma unroll
        for (int ks = 0; ks < 4; ++ks)
            bf[ks] = *(const v16bf*)(Wt + (long)col * NDIM + ks * 32 + kb2);
        const float bias = biasCat[col];

        v8f acc0 = {}, acc1 = {};
#pragma unroll
        for (int ks = 0; ks < 4; ++ks) {
            acc0 = __builtin_amdgcn_wmma_f32_16x16x32_bf16(
                     false, afrag0[ks], false, bf[ks], (short)0, acc0, false, false);
            acc1 = __builtin_amdgcn_wmma_f32_16x16x32_bf16(
                     false, afrag1[ks], false, bf[ks], (short)0, acc1, false, false);
        }
        // D layout: VGPR r -> M = r + 8*(lane>>4), N = lane&15
        float* base0 = QKV + (long)(rowBase + rsub) * QKVW + col;
        float* base1 = base0 + 16 * QKVW;
#pragma unroll
        for (int r = 0; r < 8; ++r) base0[(long)r * QKVW] = acc0[r] + bias;
#pragma unroll
        for (int r = 0; r < 8; ++r) base1[(long)r * QKVW] = acc1[r] + bias;
    }
}

// ---- kernel 4: score[e,h] = clip(dot(K[src],Q[dst])/4), atomicMax smax ---

__global__ void mha_score(const float* __restrict__ QKV, const int* __restrict__ src,
                          const int* __restrict__ dst, float* __restrict__ score,
                          unsigned* __restrict__ smax, long EH) {
    long idx = (long)blockIdx.x * blockDim.x + threadIdx.x;
    if (idx >= EH) return;
    int e  = (int)(idx >> 3);
    int hh = (int)(idx & 7);
    int s  = src[e], d0 = dst[e];
    const float4* Kp = (const float4*)(QKV + (long)s  * QKVW + HD + hh * DHEAD);
    const float4* Qp = (const float4*)(QKV + (long)d0 * QKVW + 0  + hh * DHEAD);
    float acc = 0.0f;
#pragma unroll
    for (int i = 0; i < 4; ++i) {
        float4 kv = Kp[i], qv = Qp[i];
        acc += kv.x * qv.x + kv.y * qv.y + kv.z * qv.z + kv.w * qv.w;
    }
    acc *= 0.25f;                              // 1/sqrt(16)
    acc = fminf(5.0f, fmaxf(-5.0f, acc));
    score[idx] = acc;
    atomicMax(smax + (long)d0 * NHEAD + hh, f2ord(acc));
}

// ---- kernel 5: ex = exp(score - smax[dst]); atomicAdd denom --------------

__global__ void mha_expsum(const int* __restrict__ dst, float* __restrict__ score,
                           const unsigned* __restrict__ smax, float* __restrict__ denom,
                           long EH) {
    long idx = (long)blockIdx.x * blockDim.x + threadIdx.x;
    if (idx >= EH) return;
    int e  = (int)(idx >> 3);
    int hh = (int)(idx & 7);
    int d0 = dst[e];
    float sm = ord2f(smax[(long)d0 * NHEAD + hh]);
    float ex = __expf(score[idx] - sm);
    score[idx] = ex;
    atomicAdd(denom + (long)d0 * NHEAD + hh, ex);
}

// ---- kernel 6: a = ex/denom[dst]; out[dst] += V[src] * a (atomic) --------

__global__ void mha_scatter(const float* __restrict__ QKV, const int* __restrict__ src,
                            const int* __restrict__ dst, const float* __restrict__ score,
                            const float* __restrict__ denom, float* __restrict__ out,
                            long EH) {
    long idx = (long)blockIdx.x * blockDim.x + threadIdx.x;
    if (idx >= EH) return;
    int e  = (int)(idx >> 3);
    int hh = (int)(idx & 7);
    int s  = src[e], d0 = dst[e];
    float a = score[idx] / denom[(long)d0 * NHEAD + hh];
    const float4* Vp = (const float4*)(QKV + (long)s * QKVW + 2 * HD + hh * DHEAD);
    float* op = out + (long)d0 * HD + hh * DHEAD;
#pragma unroll
    for (int i = 0; i < 4; ++i) {
        float4 v = Vp[i];
        atomicAdd(op + 4 * i + 0, v.x * a);
        atomicAdd(op + 4 * i + 1, v.y * a);
        atomicAdd(op + 4 * i + 2, v.z * a);
        atomicAdd(op + 4 * i + 3, v.w * a);
    }
}

// ---- launcher ------------------------------------------------------------

static inline size_t alignUp(size_t x, size_t a) { return (x + a - 1) & ~(a - 1); }

extern "C" void kernel_launch(void* const* d_in, const int* in_sizes, int n_in,
                              void* d_out, int out_size, void* d_ws, size_t ws_size,
                              hipStream_t stream) {
    const float* h   = (const float*)d_in[0];
    const float* Wq  = (const float*)d_in[1];
    const float* bq  = (const float*)d_in[2];
    const float* Wk  = (const float*)d_in[3];
    const float* bk  = (const float*)d_in[4];
    const float* Wv  = (const float*)d_in[5];
    const float* bv  = (const float*)d_in[6];
    const int*   src = (const int*)d_in[7];
    const int*   dst = (const int*)d_in[8];
    float* out = (float*)d_out;

    const int  n    = in_sizes[0] / NDIM;    // 50000
    const int  nBlk = (n + 255) / 256;
    const int  nPad = nBlk * 256;            // padded rows -> guard-free stores
    const long E    = in_sizes[7];           // 800000
    const long EH   = E * NHEAD;

    // workspace carve-up
    char* ws = (char*)d_ws;
    size_t off = 0;
    __bf16*   Wt      = (__bf16*)(ws + off); off = alignUp(off + (size_t)QKVW * NDIM * 2, 256);
    float*    biasCat = (float*) (ws + off); off = alignUp(off + (size_t)QKVW * 4, 256);
    float*    QKV     = (float*) (ws + off); off = alignUp(off + (size_t)nPad * QKVW * 4, 256);
    float*    score   = (float*) (ws + off); off = alignUp(off + (size_t)EH * 4, 256);
    unsigned* smax    = (unsigned*)(ws + off); off = alignUp(off + (size_t)n * NHEAD * 4, 256);
    float*    denom   = (float*) (ws + off); off = alignUp(off + (size_t)n * NHEAD * 4, 256);
    (void)ws_size; (void)n_in; (void)out_size;

    const int B = 256;
    // 1) pack weights / biases
    mha_prep<<<(QKVW * NDIM + B - 1) / B, B, 0, stream>>>(Wq, Wk, Wv, bq, bk, bv, Wt, biasCat);
    // 2) init accumulators
    long initN = (long)n * HD;
    mha_init<<<(unsigned)((initN + B - 1) / B), B, 0, stream>>>(out, denom, smax, n);
    // 3) fused QKV projection (WMMA bf16), 256 rows per block
    mha_gemm_qkv<<<nBlk, B, 0, stream>>>(h, Wt, biasCat, QKV, n);
    // 4) edge scores + segment max
    unsigned gE = (unsigned)((EH + B - 1) / B);
    mha_score<<<gE, B, 0, stream>>>(QKV, src, dst, score, smax, EH);
    // 5) exp + segment sum
    mha_expsum<<<gE, B, 0, stream>>>(dst, score, smax, denom, EH);
    // 6) normalize + weighted scatter of V
    mha_scatter<<<gE, B, 0, stream>>>(QKV, src, dst, score, denom, out, EH);
}
